// Head_52063593562737
// MI455X (gfx1250) — matrix operand order repaired and checked by reading, not compiled, not run on hardware
//
#include <hip/hip_runtime.h>

// ---------------------------------------------------------------------------
// Causal single-head attention, B=8 T=2048 C=1024 H=64, fp32 -> fp32.
// Plan (MI455X, wave32, bf16 WMMA w/ fp32 accum):
//   1) prep_w   : W (C,H) f32 -> Wt (H,C) bf16, q-scale 0.125 folded into Wq
//   2) qkv_proj : Q,K (BT,H) bf16 and Vt (B,H,T) bf16. The 192x32 bf16 weight
//                 slice (q,k,v stacked) is staged into LDS by the Tensor Data
//                 Mover (double-buffered, TENSORcnt + barrier), waves compute
//                 v_wmma_f32_16x16x32_bf16 from LDS-resident weights.
//   3) attn     : flash attention, S^T = K Q^T so softmax stats are per-lane
//                 and P lands directly in WMMA-B orientation for O^T = V^T P^T.
// Workspace: Wt 384KB | Q 2MB | K 2MB | Vt 2MB  (all L2-resident).
// ---------------------------------------------------------------------------

typedef __bf16 bf16_t;
typedef __attribute__((ext_vector_type(16))) __bf16 v16bf;
typedef __attribute__((ext_vector_type(8)))  __bf16 v8bf;
typedef __attribute__((ext_vector_type(8)))  float  v8f;
typedef __attribute__((ext_vector_type(4)))  float  v4f;
typedef __attribute__((ext_vector_type(4)))  unsigned int v4u;
typedef __attribute__((ext_vector_type(8)))  int v8i;
typedef __attribute__((ext_vector_type(4)))  int v4i;

#define DEVI __device__ __forceinline__

constexpr int Bn = 8, Tn = 2048, Cn = 1024, Hn = 64;
constexpr int BT = Bn * Tn;

DEVI v8f wmma_bf16(v16bf a, v16bf b, v8f c) {
  // D(16x16,f32) = A(16x32,bf16) * B(32x16,bf16) + C
  return __builtin_amdgcn_wmma_f32_16x16x32_bf16(
      /*neg_a=*/false, a, /*neg_b=*/false, b,
      /*c_mod=*/(short)0, c, /*reuse_a=*/false, /*reuse_b=*/false);
}

// A fragment (16x32 bf16), row-major source, per-lane row = lane&15.
// Layout: element i -> K = 16*half + i.
// p must point at &M[row][kc + 16*half].
DEVI v16bf load_afrag_bf16(const bf16_t* p) {
  union { v16bf v; struct { v8bf lo, hi; } c; } f;
  f.c.lo = *(const v8bf*)(p);
  f.c.hi = *(const v8bf*)(p + 8);
  return f.v;
}

// Same A layout but converting from f32 source on the fly.
DEVI v16bf load_afrag_f32(const float* p) {
  v4f a0 = *(const v4f*)(p);
  v4f a1 = *(const v4f*)(p + 4);
  v4f b0 = *(const v4f*)(p + 8);
  v4f b1 = *(const v4f*)(p + 12);
  v16bf r;
#pragma unroll
  for (int i = 0; i < 4; ++i) {
    r[i]      = (bf16_t)a0[i];
    r[4 + i]  = (bf16_t)a1[i];
    r[8 + i]  = (bf16_t)b0[i];
    r[12 + i] = (bf16_t)b1[i];
  }
  return r;
}

// B fragment (32x16 bf16), source is row-major B^T (per-lane col = lane&15),
// element i -> K = 16*half + i.  p must point at &Bt[col][kc + 16*half].
DEVI v16bf load_bfrag_bf16(const bf16_t* p) {
  return *(const v16bf*)p;
}

DEVI v16bf load_bfrag_f32(const float* p) {
  v16bf r;
#pragma unroll
  for (int i = 0; i < 16; i += 4) {
    v4f a = *(const v4f*)(p + i);
    r[i]     = (bf16_t)a[0];
    r[i + 1] = (bf16_t)a[1];
    r[i + 2] = (bf16_t)a[2];
    r[i + 3] = (bf16_t)a[3];
  }
  return r;
}

// Tensor Data Mover: DMA a (192 rows x 32 cols) bf16 tile of the stacked
// weight matrix Wt (192 x 1024, row stride 1024) into LDS at lds_off.
// D# built per CDNA5 ISA 8.3/8.4: group0 = count|lds_addr|global_addr|type=2,
// group1 = data_size=2B, tensor_dim0=1024, dim0_stride=1024, tensor_dim1=192,
// tile_dim0=32, tile_dim1=192. Groups 2/3 unused (2D tensor).
DEVI void tdm_load_wslice(const bf16_t* gsrc, unsigned lds_off) {
  const unsigned long long ga = (unsigned long long)(uintptr_t)gsrc;
  v4u g0;
  g0[0] = 1u;                                               // count=1 (valid D#)
  g0[1] = lds_off;                                          // lds_addr (bytes)
  g0[2] = (unsigned)(ga & 0xFFFFFFFFu);                     // global_addr[31:0]
  g0[3] = (unsigned)((ga >> 32) & 0x1FFFFFFu) | (2u << 30); // addr[56:32]|type=2
  v8i g1;
  g1[0] = 0x00010000;            // workgroup_mask=0, data_size=1 (2 bytes)
  g1[1] = (int)(1024u << 16);    // tensor_dim0 = 1024 (bits 79:48, low part)
  g1[2] = (int)(192u << 16);     // tensor_dim0 hi=0 | tensor_dim1 = 192
  g1[3] = (int)(32u << 16);      // tensor_dim1 hi=0 | tile_dim0 = 32
  g1[4] = 192;                   // tile_dim1 = 192, tile_dim2 = 0
  g1[5] = 1024;                  // tensor_dim0_stride = 1024 (low 32)
  g1[6] = 0;                     // stride0 hi | stride1 lo
  g1[7] = 0;                     // stride1 hi
  v4i z4 = {0, 0, 0, 0};
#if __clang_major__ >= 23
  v8i z8 = {0, 0, 0, 0, 0, 0, 0, 0};
  __builtin_amdgcn_tensor_load_to_lds(g0, g1, z4, z4, z8, 0);
#else
  __builtin_amdgcn_tensor_load_to_lds(g0, g1, z4, z4, 0);
#endif
}

// --------------------------- kernel 1: weights -----------------------------
__global__ __launch_bounds__(256) void prep_w(
    const float* __restrict__ Wk, const float* __restrict__ Wq,
    const float* __restrict__ Wv, bf16_t* __restrict__ Wt) {
  int idx = blockIdx.x * blockDim.x + threadIdx.x;   // over H*C
  if (idx >= Hn * Cn) return;
  int h = idx / Cn, c = idx % Cn;
  // Wt[w][h][c] = W[c][h];  order: 0=q (scaled), 1=k, 2=v
  Wt[(size_t)0 * Hn * Cn + idx] = (bf16_t)(Wq[(size_t)c * Hn + h] * 0.125f);
  Wt[(size_t)1 * Hn * Cn + idx] = (bf16_t)(Wk[(size_t)c * Hn + h]);
  Wt[(size_t)2 * Hn * Cn + idx] = (bf16_t)(Wv[(size_t)c * Hn + h]);
}

// ------------------------ kernel 2: QKV projection --------------------------
// One wave owns 16 rows of x. All 8 waves of the block share TDM-staged weight
// slices in LDS (double-buffered). Q,K tiles computed transposed (A=Wt, B=x)
// so stores to row-major (T,H) are contiguous; V computed direct (A=x, B=Wt)
// so stores to Vt (H,T) are contiguous.
__global__ __launch_bounds__(256) void qkv_proj(
    const float* __restrict__ x, const bf16_t* __restrict__ Wt,
    bf16_t* __restrict__ Q, bf16_t* __restrict__ K, bf16_t* __restrict__ Vt) {
  __shared__ bf16_t smem[2][3 * Hn * 32];        // 2 x 12KB weight slices

  const int wave = threadIdx.x >> 5;             // 0..7
  const int lane = threadIdx.x & 31;
  const int r16  = lane & 15;
  const int hf   = lane >> 4;
  const int t0   = (blockIdx.x * 8 + wave) * 16; // 16-row block of x
  if (t0 >= BT) return;

  const float* xA = x + (size_t)(t0 + r16) * Cn + 16 * hf;   // A-frag base
  const float* xB = x + (size_t)(t0 + r16) * Cn + 16 * hf;   // B-frag base

  // Kick off the first weight slice DMA (wave 0 owns the TDM + TENSORcnt).
  if (wave == 0) {
    tdm_load_wslice(Wt, (unsigned)(uintptr_t)&smem[0][0]);
  }

  v8f acc[12];  // [0..3]=Q htiles, [4..7]=K htiles, [8..11]=V htiles
#pragma unroll
  for (int i = 0; i < 12; ++i) acc[i] = v8f{};

  for (int kc = 0; kc < Cn; kc += 32) {
    const int p = (kc >> 5) & 1;
    if (wave == 0) __builtin_amdgcn_s_wait_tensorcnt(0);   // slice p landed
    __syncthreads();                                       // publish buffer p
    if (wave == 0 && kc + 32 < Cn) {                       // prefetch slice p^1
      tdm_load_wslice(Wt + (size_t)(kc + 32), (unsigned)(uintptr_t)&smem[p ^ 1][0]);
    }

    v16bf xfB = load_bfrag_f32(xB + kc);   // x as WMMA-B (for Q,K)
    v16bf xfA = load_afrag_f32(xA + kc);   // x as WMMA-A (for V)
    const bf16_t* sm = &smem[p][0];

    // Q, K: D = Wt_tile * x_tile -> D(m=h, n=t); Wt rows from LDS (A layout)
#pragma unroll
    for (int w = 0; w < 2; ++w) {
#pragma unroll
      for (int ht = 0; ht < 4; ++ht) {
        v16bf wf = load_afrag_bf16(sm + (w * Hn + ht * 16 + r16) * 32 + 16 * hf);
        acc[w * 4 + ht] = wmma_bf16(wf, xfB, acc[w * 4 + ht]);
      }
    }
    // V: D = x_tile * Wt_tile -> D(m=t, n=h); Wt rows from LDS (B layout)
#pragma unroll
    for (int ht = 0; ht < 4; ++ht) {
      v16bf wf = load_bfrag_bf16(sm + (2 * Hn + ht * 16 + r16) * 32 + 16 * hf);
      acc[8 + ht] = wmma_bf16(xfA, wf, acc[8 + ht]);
    }
    __syncthreads();   // all waves done with buffer p before it is refilled
  }

  // ---- stores
#pragma unroll
  for (int w = 0; w < 2; ++w) {
    bf16_t* dst = (w == 0) ? Q : K;
#pragma unroll
    for (int ht = 0; ht < 4; ++ht) {
      v8bf pk;
#pragma unroll
      for (int r = 0; r < 8; ++r) pk[r] = (bf16_t)acc[w * 4 + ht][r];
      // lane: t = t0+r16, vgpr r: h = ht*16 + 8*hf + r  (8 contiguous)
      *(v8bf*)(dst + (size_t)(t0 + r16) * Hn + ht * 16 + 8 * hf) = pk;
    }
  }
  {
    const int b = t0 / Tn, tl = t0 % Tn;
#pragma unroll
    for (int ht = 0; ht < 4; ++ht) {
      v8bf pk;
#pragma unroll
      for (int r = 0; r < 8; ++r) pk[r] = (bf16_t)acc[8 + ht][r];
      // lane: h = ht*16 + r16, vgpr r: t = tl + 8*hf + r (8 contiguous)
      *(v8bf*)(Vt + ((size_t)b * Hn + ht * 16 + r16) * Tn + tl + 8 * hf) = pk;
    }
  }
}

// ------------------------ kernel 3: flash attention -------------------------
__global__ __launch_bounds__(256) void attn(
    const bf16_t* __restrict__ Q, const bf16_t* __restrict__ K,
    const bf16_t* __restrict__ Vt, float* __restrict__ out) {
  const int wave = (blockIdx.x * blockDim.x + threadIdx.x) >> 5;
  const int lane = threadIdx.x & 31;
  const int r16  = lane & 15;
  const int hf   = lane >> 4;
  if (wave >= BT / 16) return;
  const int b  = (wave * 16) / Tn;
  const int t0 = (wave * 16) % Tn;

  const bf16_t* Qb = Q  + (size_t)b * Tn * Hn;
  const bf16_t* Kb = K  + (size_t)b * Tn * Hn;
  const bf16_t* Vb = Vt + (size_t)b * Hn * Tn;

  // Q as WMMA-B fragments (n = t), k = head dim: chunks h=0..31 and h=32..63.
  v16bf qf0 = load_bfrag_bf16(Qb + (size_t)(t0 + r16) * Hn + 0  + 16 * hf);
  v16bf qf1 = load_bfrag_bf16(Qb + (size_t)(t0 + r16) * Hn + 32 + 16 * hf);

  v8f accO[4] = {v8f{}, v8f{}, v8f{}, v8f{}};
  float m_i = -1e30f, l_i = 0.f;
  const int tq = t0 + r16;           // this lane's query index

  for (int s0 = 0; s0 < t0 + 16; s0 += 32) {
    __builtin_prefetch(Kb + (size_t)(s0 + 32 + r16) * Hn, 0, 1);
    __builtin_prefetch(Vb + (size_t)r16 * Tn + s0 + 32, 0, 1);

    // S^T = K * Q^T : two 16(key)x16(query) tiles, K-dim = 64 head dims.
    v16bf kf00 = load_afrag_bf16(Kb + (size_t)(s0 + r16) * Hn + 0  + 16 * hf);
    v16bf kf01 = load_afrag_bf16(Kb + (size_t)(s0 + r16) * Hn + 32 + 16 * hf);
    v16bf kf10 = load_afrag_bf16(Kb + (size_t)(s0 + 16 + r16) * Hn + 0  + 16 * hf);
    v16bf kf11 = load_afrag_bf16(Kb + (size_t)(s0 + 16 + r16) * Hn + 32 + 16 * hf);
    v8f c0 = v8f{}; c0 = wmma_bf16(kf00, qf0, c0); c0 = wmma_bf16(kf01, qf1, c0);
    v8f c1 = v8f{}; c1 = wmma_bf16(kf10, qf0, c1); c1 = wmma_bf16(kf11, qf1, c1);

    // Causal mask + block max.  vgpr r of c0 -> key s0 + 8*hf + r; c1 -> +16.
    float mb = -1e30f;
#pragma unroll
    for (int r = 0; r < 8; ++r) {
      const int sa = s0 + 8 * hf + r;
      c0[r] = (sa      > tq) ? -1e30f : c0[r];
      c1[r] = (sa + 16 > tq) ? -1e30f : c1[r];
      mb = fmaxf(mb, fmaxf(c0[r], c1[r]));
    }
    mb = fmaxf(mb, __shfl_xor(mb, 16, 32));          // combine half-waves
    const float m_new = fmaxf(m_i, mb);
    const float corr  = __expf(m_i - m_new);

    // exp + pack P into WMMA-B orientation (element i -> key 16*hf + i).
    float ls = 0.f;
    v16bf pf;
#pragma unroll
    for (int r = 0; r < 8; ++r) {
      const float p0 = __expf(c0[r] - m_new);        // key 8*hf + r
      const float p1 = __expf(c1[r] - m_new);        // key 16 + 8*hf + r
      ls += p0 + p1;
      const float q0 = __shfl_xor(p0, 16, 32);       // partner half's p0
      const float q1 = __shfl_xor(p1, 16, 32);       // partner half's p1
      pf[r]     = (bf16_t)(hf ? q1 : p0);            // K = 16*hf + r
      pf[8 + r] = (bf16_t)(hf ? p1 : q0);            // K = 16*hf + 8 + r
    }
    ls += __shfl_xor(ls, 16, 32);
    l_i = l_i * corr + ls;
    m_i = m_new;

#pragma unroll
    for (int ht = 0; ht < 4; ++ht)
#pragma unroll
      for (int r = 0; r < 8; ++r) accO[ht][r] *= corr;

    // O^T += V^T * P^T : A = Vt tile (m=h, k=key), B = pf.
#pragma unroll
    for (int ht = 0; ht < 4; ++ht) {
      v16bf vf = load_afrag_bf16(Vb + (size_t)(ht * 16 + r16) * Tn + s0 + 16 * hf);
      accO[ht] = wmma_bf16(vf, pf, accO[ht]);
    }
  }

  // Normalize and store: lane t = t0+r16; vgpr r -> h = ht*16 + 8*hf + r.
  const float inv = 1.f / l_i;
  float* orow = out + ((size_t)b * Tn + t0 + r16) * Hn;
#pragma unroll
  for (int ht = 0; ht < 4; ++ht) {
    v4f o0, o1;
#pragma unroll
    for (int r = 0; r < 4; ++r) {
      o0[r] = accO[ht][r] * inv;
      o1[r] = accO[ht][4 + r] * inv;
    }
    *(v4f*)(orow + ht * 16 + 8 * hf)     = o0;
    *(v4f*)(orow + ht * 16 + 8 * hf + 4) = o1;
  }
}

// ------------------------------- launcher -----------------------------------
extern "C" void kernel_launch(void* const* d_in, const int* in_sizes, int n_in,
                              void* d_out, int out_size, void* d_ws, size_t ws_size,
                              hipStream_t stream) {
  const float* x  = (const float*)d_in[0];
  const float* Wk = (const float*)d_in[1];
  const float* Wq = (const float*)d_in[2];
  const float* Wv = (const float*)d_in[3];
  float* out = (float*)d_out;

  char* ws = (char*)d_ws;
  bf16_t* Wt = (bf16_t*)(ws);                                   // 3*H*C  = 384 KB
  bf16_t* Q  = (bf16_t*)(ws + (512u << 10));                    // BT*H   = 2 MB
  bf16_t* K  = (bf16_t*)(ws + (512u << 10) + (2u << 20));       // BT*H   = 2 MB
  bf16_t* Vt = (bf16_t*)(ws + (512u << 10) + (4u << 20));       // B*H*T  = 2 MB

  prep_w<<<(Hn * Cn + 255) / 256, 256, 0, stream>>>(Wk, Wq, Wv, Wt);
  qkv_proj<<<(BT / 16) / 8, 256, 0, stream>>>(x, Wt, Q, K, Vt);
  attn<<<(BT / 16) / 8, 256, 0, stream>>>(Q, K, Vt, out);
}
